// ECGGraphNetwork_2963527434774
// MI455X (gfx1250) — matrix operand order repaired and checked by reading, not compiled, not run on hardware
//
#include <hip/hip_runtime.h>
#include <hip/hip_bf16.h>

// ---------------------------------------------------------------------------
// Fused ECG graph network for MI455X (gfx1250), wave32 + WMMA bf16 + TDM.
// One workgroup (256 threads = 8 waves) processes NB=8 batches (96 rows)
// entirely in LDS: GCN1 -> GCN2 -> MHA -> LN -> GCN3 -> pooling.
// X chunks are DMA'd into LDS by the Tensor Data Mover (TENSORcnt).
// ---------------------------------------------------------------------------

#define NB     8            // batches per workgroup
#define SEQ    12           // leads
#define MROWS  96           // NB*SEQ
#define EMB    256          // hidden size
#define FIN    512          // input features
#define KC     64           // K chunk staged in LDS per GEMM step
#define AST    264          // activation row stride (elems, 528B = 16B-mult, bank-safe)
#define KST    72           // B^T row stride (elems, 144B = 16B-mult)
#define XST    68           // f32 X chunk row stride in DWORDs (272B, TDM-padded)

typedef __attribute__((ext_vector_type(16))) __bf16 v16bf;
typedef __attribute__((ext_vector_type(8)))  __bf16 v8bf;
typedef __attribute__((ext_vector_type(8)))  float  v8f;
typedef __attribute__((ext_vector_type(4)))  float  v4f;
typedef __attribute__((ext_vector_type(8)))  unsigned short v8u;
typedef __attribute__((ext_vector_type(4)))  unsigned int v4u32;
typedef __attribute__((ext_vector_type(8)))  int v8i32;
typedef __attribute__((ext_vector_type(4)))  int v4i32;

// fixed 12-lead adjacency (with self loops) as row bitmasks + degrees
__constant__ unsigned short AMASK[12] = {95u, 575u, 2095u, 15u, 19u, 38u,
                                         193u, 448u, 896u, 1794u, 3584u, 3076u};
__constant__ float ADEG[12] = {6.f,7.f,6.f,4.f,3.f,3.f,3.f,3.f,3.f,4.f,3.f,3.f};

// native bf16 converts (v_cvt_bf16_f32 / v_cvt_pk_bf16_f32, RNE)
__device__ __forceinline__ unsigned short f2bf(float f){
  return __builtin_bit_cast(unsigned short, (__bf16)f);
}
__device__ __forceinline__ float bf2f(unsigned short h){
  return (float)__builtin_bit_cast(__bf16, h);
}
__device__ __forceinline__ v8f zero8(){
  v8f z;
#pragma unroll
  for (int i = 0; i < 8; ++i) z[i] = 0.0f;
  return z;
}
__device__ __forceinline__ v16bf cat8(v8bf lo, v8bf hi){
  return __builtin_shufflevector(lo, hi, 0,1,2,3,4,5,6,7,8,9,10,11,12,13,14,15);
}

// A fragment (16x32 bf16, MxK): lanes 0-15 rows, half-wave selects K 0-7/16-23
// vs 8-15/24-31; element order = ascending K within each 16B run.
__device__ __forceinline__ v16bf load_fragA(const unsigned short* base, int stride,
                                            int row0, int k0, int lane){
  int r    = row0 + (lane & 15);
  int half = lane >> 4;
  const unsigned short* p = base + r * stride + k0 + 8 * half;
  v8bf lo = __builtin_bit_cast(v8bf, *(const v8u*)p);        // K = k0+8h .. +7
  v8bf hi = __builtin_bit_cast(v8bf, *(const v8u*)(p + 16)); // K = k0+16+8h .. +7
  return cat8(lo, hi);
}

// Same fragment but sourced from a TDM-loaded f32 tile (row pitch XST dwords),
// converting f32 -> bf16 via packed converts.
__device__ __forceinline__ v16bf load_fragA_f32(const float* base, int row0, int k0,
                                                int lane){
  int r    = row0 + (lane & 15);
  int half = lane >> 4;
  const float* p = base + r * XST + k0 + 8 * half;
  v4f a0 = *(const v4f*)p;
  v4f a1 = *(const v4f*)(p + 4);
  v4f b0 = *(const v4f*)(p + 16);
  v4f b1 = *(const v4f*)(p + 20);
  v8f lo = __builtin_shufflevector(a0, a1, 0,1,2,3,4,5,6,7);
  v8f hi = __builtin_shufflevector(b0, b1, 0,1,2,3,4,5,6,7);
  return cat8(__builtin_convertvector(lo, v8bf), __builtin_convertvector(hi, v8bf));
}

// B fragment (32x16 bf16, KxN) from LDS staged as B^T[n][k].
__device__ __forceinline__ v16bf load_fragB(const unsigned short* bt,
                                            int n0, int k0, int lane){
  int c    = n0 + (lane & 15);
  int kseg = (lane >> 4) << 3;
  const unsigned short* p = bt + c * KST + k0 + kseg;
  v8bf lo = __builtin_bit_cast(v8bf, *(const v8u*)p);
  v8bf hi = __builtin_bit_cast(v8bf, *(const v8u*)(p + 16));
  return cat8(lo, hi);
}

__device__ __forceinline__ v8f wmma_bf16(v16bf a, v16bf b, v8f c){
  return __builtin_amdgcn_wmma_f32_16x16x32_bf16(false, a, false, b,
                                                 (short)0, c, false, false);
}

// 3x4 tile block per wave: 3 A-frags + 4 B-frags feed 12 WMMAs (reuse).
__device__ __forceinline__ void mma_block(v8f acc[12], const v16bf af[3],
                                          const v16bf bfv[4]){
#pragma unroll
  for (int mi = 0; mi < 3; ++mi)
#pragma unroll
    for (int ni = 0; ni < 4; ++ni)
      acc[mi * 4 + ni] = wmma_bf16(af[mi], bfv[ni], acc[mi * 4 + ni]);
}

// TDM: DMA one f32 X chunk [96 rows x 64 cols, row stride 512] into LDS with
// 4-DWORD padding every 64 DWORDs (row pitch 68 dwords). D# per ISA 08 §8.
__device__ __forceinline__ void tdm_load_x_chunk(const float* gsrc, unsigned lds_off){
  unsigned long long ga = (unsigned long long)(uintptr_t)gsrc;
  v4u32 g0;
  g0[0] = 1u;                                            // count=1 (valid user D#)
  g0[1] = lds_off;                                       // lds_addr (bytes)
  g0[2] = (unsigned)(ga & 0xFFFFFFFFull);                // global_addr lo
  g0[3] = (unsigned)((ga >> 32) & 0x01FFFFFFull) | (2u << 30); // addr hi | type=2
  v8i32 g1;
  g1[0] = (int)((2u << 16)        // data_size = 4B
              | (1u << 20)        // pad_enable
              | (5u << 22)        // pad_interval: 64 DWORDs
              | (3u << 25));      // pad_amount: 4 DWORDs
  g1[1] = (int)(512u << 16);                // tensor_dim0 = 512 (lo16)
  g1[2] = (int)(96u << 16);                 // tdim0 hi=0 | tensor_dim1 = 96 (lo16)
  g1[3] = (int)(64u << 16);                 // tdim1 hi=0 | tile_dim0 = 64
  g1[4] = (int)96u;                         // tile_dim1 = 96, tile_dim2 = 0
  g1[5] = (int)512u;                        // tensor_dim0_stride = 512 (lo32)
  g1[6] = 0;                                // stride hi | dim1_stride lo (unused, 2D)
  g1[7] = 0;
  v4i32 z4 = {0, 0, 0, 0};                  // groups 2/3 unused (2D tensor)
  v8i32 z8 = {0, 0, 0, 0, 0, 0, 0, 0};      // extra SGPR group (clang-23 form)
  __builtin_amdgcn_tensor_load_to_lds(g0, g1, z4, z4, z8, 0);
}

// Stage one KCxEMB weight chunk into LDS as B^T (bf16).
__device__ __forceinline__ void stageBT(unsigned short* bt, const float* Wg, int ldw,
                                        int kc0, int noff, int transB, int tid){
  if (transB){
    for (int e = tid; e < KC * EMB; e += 256){
      int n = e >> 6, k = e & 63;                       // coalesced over k
      bt[n * KST + k] = f2bf(Wg[(noff + n) * ldw + (kc0 + k)]);
    }
  } else {
    for (int e = tid; e < KC * EMB; e += 256){
      int k = e >> 8, n = e & 255;                      // coalesced over n
      bt[n * KST + k] = f2bf(Wg[(kc0 + k) * ldw + n]);
    }
  }
}

// Generic GEMM: acc[96x256] += A_lds[96xK](bf16, stride AST) * W (staged per chunk).
__device__ __forceinline__ void gemm_lds(v8f acc[12], const unsigned short* Alds,
                                         unsigned short* sBT, const float* Wg, int ldw,
                                         int noff, int transB, int K,
                                         int mw, int nw, int tid){
  const int lane = tid & 31;
#pragma unroll
  for (int t = 0; t < 12; ++t) acc[t] = zero8();
  for (int kc0 = 0; kc0 < K; kc0 += KC){
    __syncthreads();                       // previous chunk fully consumed
    if (kc0 + KC < K){                     // L2 prefetch hint for next chunk
      const float* nx = transB ? (Wg + (size_t)noff * ldw + (kc0 + KC))
                               : (Wg + (size_t)(kc0 + KC) * ldw);
      __builtin_prefetch(nx + (tid << 5), 0, 1);
    }
    stageBT(sBT, Wg, ldw, kc0, noff, transB, tid);
    __syncthreads();                       // chunk staged
#pragma unroll
    for (int kk = 0; kk < KC; kk += 32){
      v16bf af[3], bfv[4];
#pragma unroll
      for (int mi = 0; mi < 3; ++mi)
        af[mi] = load_fragA(Alds, AST, (mw * 3 + mi) << 4, kc0 + kk, lane);
#pragma unroll
      for (int ni = 0; ni < 4; ++ni)
        bfv[ni] = load_fragB(sBT, (nw * 4 + ni) << 4, kk, lane);
      mma_block(acc, af, bfv);
    }
  }
}

// C/D layout: lanes 0-15 cols, VGPR v -> row v (lanes<16) / 8+v (lanes>=16).
__device__ __forceinline__ void wb_raw(unsigned short* dst, const v8f acc[12],
                                       int mw, int nw, int lane){
#pragma unroll
  for (int mi = 0; mi < 3; ++mi){
    int rb = ((mw * 3 + mi) << 4) + ((lane >> 4) << 3);
#pragma unroll
    for (int ni = 0; ni < 4; ++ni){
      int col = ((nw * 4 + ni) << 4) + (lane & 15);
#pragma unroll
      for (int v = 0; v < 8; ++v)
        dst[(rb + v) * AST + col] = f2bf(acc[mi * 4 + ni][v]);
    }
  }
}
__device__ __forceinline__ void wb_bias(unsigned short* dst, const v8f acc[12],
                                        const float* biasg, int boff,
                                        int mw, int nw, int lane){
#pragma unroll
  for (int mi = 0; mi < 3; ++mi){
    int rb = ((mw * 3 + mi) << 4) + ((lane >> 4) << 3);
#pragma unroll
    for (int ni = 0; ni < 4; ++ni){
      int col = ((nw * 4 + ni) << 4) + (lane & 15);
      float bb = biasg[boff + col];
#pragma unroll
      for (int v = 0; v < 8; ++v)
        dst[(rb + v) * AST + col] = f2bf(acc[mi * 4 + ni][v] + bb);
    }
  }
}

// GCN normalization: dst[b,i,:] = act(bias + sum_j Anorm[i,j]*src[b,j,:])
__device__ __forceinline__ void mix_rows(unsigned short* dst, const unsigned short* src,
                                         const float* biasg, int relu,
                                         const float* sA12, int tid){
  for (int e = tid; e < MROWS * EMB; e += 256){
    int c = e & 255, rr = e >> 8;
    int b = rr / SEQ, i = rr - b * SEQ;
    const float* Ar = sA12 + i * SEQ;
    int base = b * SEQ;
    float a = biasg[c];
#pragma unroll
    for (int j = 0; j < SEQ; ++j)
      a += Ar[j] * bf2f(src[(base + j) * AST + c]);
    if (relu) a = fmaxf(a, 0.0f);
    dst[rr * AST + c] = f2bf(a);
  }
}

__global__ __launch_bounds__(256) void ecg_fused_kernel(
    const float* __restrict__ Xg,
    const float* __restrict__ W1, const float* __restrict__ B1,
    const float* __restrict__ W2, const float* __restrict__ B2,
    const float* __restrict__ W3, const float* __restrict__ B3,
    const float* __restrict__ INW, const float* __restrict__ INB,
    const float* __restrict__ OUTW, const float* __restrict__ OUTB,
    const float* __restrict__ LNG, const float* __restrict__ LNB,
    float* __restrict__ Og)
{
  __shared__ unsigned short sHA[MROWS * AST];   // current activation (bf16)
  __shared__ unsigned short sTF[MROWS * AST];   // raw GEMM out / TDM X chunk / attn O
  __shared__ unsigned short sQ [MROWS * AST];   // Q, later att, later h3
  __shared__ unsigned short sK [MROWS * AST];
  __shared__ unsigned short sV [MROWS * AST];
  __shared__ unsigned short sBT[EMB * KST];     // staged B^T chunk
  __shared__ float sA12[SEQ * SEQ];
  __shared__ float sMean[MROWS];
  __shared__ float sWatt[NB * SEQ];

  const int tid   = threadIdx.x;
  const int lane  = tid & 31;
  const int wave  = tid >> 5;
  const int mw    = wave & 1;                   // wave's m-block (0..1) of 3 tiles
  const int nw    = wave >> 1;                  // wave's n-block (0..3) of 4 tiles
  const int rowg0 = blockIdx.x * MROWS;

  if (tid < SEQ * SEQ){
    int i = tid / SEQ, j = tid - i * SEQ;
    sA12[tid] = ((AMASK[i] >> j) & 1) ? rsqrtf(ADEG[i] * ADEG[j]) : 0.0f;
  }

  v8f acc[12];

  // ---------------- GCN layer 1: X[96,512] @ W1 (X via TDM) ----------------
  float* sXF = (float*)sTF;                     // alias: f32 X chunk [96][68]
  const unsigned xlds = (unsigned)(uintptr_t)sXF;  // LDS byte offset (addr[31:0])
#pragma unroll
  for (int t = 0; t < 12; ++t) acc[t] = zero8();
  for (int kc0 = 0; kc0 < FIN; kc0 += KC){
    __syncthreads();                            // previous chunk consumed
    if (wave == 0)
      tdm_load_x_chunk(Xg + (size_t)rowg0 * FIN + kc0, xlds);
    if (kc0 + KC < FIN)
      __builtin_prefetch(W1 + (size_t)(kc0 + KC) * EMB + (tid << 5), 0, 1);
    stageBT(sBT, W1, EMB, kc0, 0, 0, tid);      // overlaps with TDM in flight
    if (wave == 0)
      __builtin_amdgcn_s_wait_tensorcnt(0);
    __syncthreads();                            // X tile + B^T chunk visible
#pragma unroll
    for (int kk = 0; kk < KC; kk += 32){
      v16bf af[3], bfv[4];
#pragma unroll
      for (int mi = 0; mi < 3; ++mi)
        af[mi] = load_fragA_f32(sXF, (mw * 3 + mi) << 4, kk, lane);
#pragma unroll
      for (int ni = 0; ni < 4; ++ni)
        bfv[ni] = load_fragB(sBT, (nw * 4 + ni) << 4, kk, lane);
      mma_block(acc, af, bfv);
    }
  }
  __syncthreads();                              // X-chunk reads done (aliases sTF)
  wb_raw(sTF, acc, mw, nw, lane);
  __syncthreads();
  mix_rows(sHA, sTF, B1, 1, sA12, tid);         // A-norm + bias + relu -> H1

  // ---------------- GCN layer 2 ----------------
  gemm_lds(acc, sHA, sBT, W2, EMB, 0, 0, EMB, mw, nw, tid);
  __syncthreads();
  wb_raw(sTF, acc, mw, nw, lane);
  __syncthreads();
  mix_rows(sHA, sTF, B2, 1, sA12, tid);         // -> H2 (kept for residual)

  // ---------------- MHA: QKV projections (x @ in_w^T + in_b) ----------------
  gemm_lds(acc, sHA, sBT, INW, EMB, 0,       1, EMB, mw, nw, tid);
  __syncthreads();
  wb_bias(sQ, acc, INB, 0,       mw, nw, lane);
  gemm_lds(acc, sHA, sBT, INW, EMB, EMB,     1, EMB, mw, nw, tid);
  __syncthreads();
  wb_bias(sK, acc, INB, EMB,     mw, nw, lane);
  gemm_lds(acc, sHA, sBT, INW, EMB, 2 * EMB, 1, EMB, mw, nw, tid);
  __syncthreads();
  wb_bias(sV, acc, INB, 2 * EMB, mw, nw, lane);
  __syncthreads();

  // ---------------- attention core (12x12 per batch-head, VALU) -------------
  for (int p = wave; p < NB * 4; p += 8){
    int b = p >> 2, h = p & 3;
    int r0 = b * SEQ, c0 = h * 64;
    if (lane < SEQ){
      float sc[SEQ];
      float mx = -1e30f;
      const unsigned short* qp = sQ + (r0 + lane) * AST + c0;
#pragma unroll
      for (int j = 0; j < SEQ; ++j){
        const unsigned short* kp = sK + (r0 + j) * AST + c0;
        float s = 0.0f;
        for (int d = 0; d < 64; ++d) s += bf2f(qp[d]) * bf2f(kp[d]);
        sc[j] = s * 0.125f;                     // 1/sqrt(64)
        mx = fmaxf(mx, sc[j]);
      }
      float den = 0.0f;
#pragma unroll
      for (int j = 0; j < SEQ; ++j){ sc[j] = __expf(sc[j] - mx); den += sc[j]; }
      float inv = 1.0f / den;
      for (int d = 0; d < 64; ++d){
        float o = 0.0f;
#pragma unroll
        for (int j = 0; j < SEQ; ++j) o += sc[j] * bf2f(sV[(r0 + j) * AST + c0 + d]);
        sTF[(r0 + lane) * AST + c0 + d] = f2bf(o * inv);   // O tile
      }
    }
  }

  // ---------------- out_proj: O @ out_w^T + out_b ----------------
  gemm_lds(acc, sTF, sBT, OUTW, EMB, 0, 1, EMB, mw, nw, tid);
  __syncthreads();
  wb_bias(sQ, acc, OUTB, 0, mw, nw, lane);      // att result
  __syncthreads();

  // ---------------- residual + LayerNorm -> sHA ----------------
  for (int r = wave; r < MROWS; r += 8){
    float s = 0.0f, s2 = 0.0f;
    for (int c = lane; c < EMB; c += 32){
      float v = bf2f(sHA[r * AST + c]) + bf2f(sQ[r * AST + c]);
      s += v; s2 += v * v;
    }
#pragma unroll
    for (int m = 16; m; m >>= 1){
      s  += __shfl_xor(s,  m, 32);
      s2 += __shfl_xor(s2, m, 32);
    }
    float mu  = s  * (1.0f / EMB);
    float var = s2 * (1.0f / EMB) - mu * mu;
    float rs  = rsqrtf(var + 1e-5f);
    for (int c = lane; c < EMB; c += 32){
      float v = bf2f(sHA[r * AST + c]) + bf2f(sQ[r * AST + c]);
      sHA[r * AST + c] = f2bf((v - mu) * rs * LNG[c] + LNB[c]);
    }
  }

  // ---------------- GCN layer 3 (no relu) -> h3 in sQ ----------------
  gemm_lds(acc, sHA, sBT, W3, EMB, 0, 0, EMB, mw, nw, tid);
  __syncthreads();
  wb_raw(sTF, acc, mw, nw, lane);
  __syncthreads();
  mix_rows(sQ, sTF, B3, 0, sA12, tid);
  __syncthreads();

  // ---------------- attention pooling + max pooling ----------------
  for (int r = wave; r < MROWS; r += 8){
    float s = 0.0f;
    for (int c = lane; c < EMB; c += 32) s += bf2f(sQ[r * AST + c]);
#pragma unroll
    for (int m = 16; m; m >>= 1) s += __shfl_xor(s, m, 32);
    if (lane == 0) sMean[r] = s * (1.0f / EMB);
  }
  __syncthreads();
  if (tid < NB){
    float mx = -1e30f;
#pragma unroll
    for (int i = 0; i < SEQ; ++i) mx = fmaxf(mx, sMean[tid * SEQ + i]);
    float den = 0.0f, ex[SEQ];
#pragma unroll
    for (int i = 0; i < SEQ; ++i){ ex[i] = __expf(sMean[tid * SEQ + i] - mx); den += ex[i]; }
    float inv = 1.0f / den;
#pragma unroll
    for (int i = 0; i < SEQ; ++i) sWatt[tid * SEQ + i] = ex[i] * inv;
  }
  __syncthreads();
  for (int e = tid; e < NB * EMB; e += 256){
    int c = e & 255, b = e >> 8;
    float ws = 0.0f, mx = -1e30f;
#pragma unroll
    for (int i = 0; i < SEQ; ++i){
      float v = bf2f(sQ[(b * SEQ + i) * AST + c]);
      ws += v * sWatt[b * SEQ + i];
      mx  = fmaxf(mx, v);
    }
    int gb = blockIdx.x * NB + b;
    Og[gb * (2 * EMB) + c]        = ws;   // x_weighted
    Og[gb * (2 * EMB) + EMB + c]  = mx;   // x_max
  }
}

extern "C" void kernel_launch(void* const* d_in, const int* in_sizes, int n_in,
                              void* d_out, int out_size, void* d_ws, size_t ws_size,
                              hipStream_t stream) {
  (void)n_in; (void)out_size; (void)d_ws; (void)ws_size;
  const float* x    = (const float*)d_in[0];
  const float* W1   = (const float*)d_in[1];
  const float* b1   = (const float*)d_in[2];
  const float* W2   = (const float*)d_in[3];
  const float* b2   = (const float*)d_in[4];
  const float* W3   = (const float*)d_in[5];
  const float* b3   = (const float*)d_in[6];
  const float* inw  = (const float*)d_in[7];
  const float* inb  = (const float*)d_in[8];
  const float* outw = (const float*)d_in[9];
  const float* outb = (const float*)d_in[10];
  const float* lng  = (const float*)d_in[11];
  const float* lnb  = (const float*)d_in[12];
  float* out = (float*)d_out;

  const int B = in_sizes[0] / (SEQ * FIN);     // 8192
  dim3 grid(B / NB), block(256);
  ecg_fused_kernel<<<grid, block, 0, stream>>>(x, W1, b1, W2, b2, W3, b3,
                                               inw, inb, outw, outb, lng, lnb, out);
}
